// denoising_78786880078249
// MI455X (gfx1250) — matrix operand order repaired and checked by reading, not compiled, not run on hardware
//
#include <hip/hip_runtime.h>
#include <hip/hip_bf16.h>
#include <stdint.h>

#define N_    16
#define C_    64
#define CH_   32
#define S_    3136          // 56*56
#define QT_   196           // S_/16 query tiles per batch
#define NT_   (N_*QT_)
#define NCS_  (N_*C_*S_)    // 3,211,264
#define NKCH_ 98            // key chunks of 32
#define WAVES_ 7
#define BLK2_ (WAVES_*32)
#define BN_EPS_ 1e-5f
#define SCALE_ 0.17677669529663687f   // 1/sqrt(32)

typedef __bf16 bf16_t;
typedef float  v8f   __attribute__((ext_vector_type(8)));
typedef __bf16 v16bf __attribute__((ext_vector_type(16)));
typedef unsigned int v4u __attribute__((ext_vector_type(4)));
typedef int v8i __attribute__((ext_vector_type(8)));
typedef int v4i __attribute__((ext_vector_type(4)));

#if defined(__AMDGCN__) && __has_builtin(__builtin_amdgcn_tensor_load_to_lds) && __has_builtin(__builtin_amdgcn_s_wait_tensorcnt)
#define HAVE_TDM 1
#else
#define HAVE_TDM 0
#endif

#if defined(__AMDGCN__) && __has_builtin(__builtin_amdgcn_permlane16)
#define HAVE_PERMLANE 1
#else
#define HAVE_PERMLANE 0
#endif

union FragU { uint4 u[2]; v16bf v; };

// ---- xor-exchange within each 16-lane half (VALU v_permlane16, no LDS) -------
__host__ __device__ constexpr unsigned selpart(int m, int base) {
  unsigned v = 0;
  for (int j = 0; j < 8; ++j) v |= ((unsigned)((base + j) ^ m) & 0xFu) << (4 * j);
  return v;
}

template <int M>
__device__ __forceinline__ float xlane16(float x) {
#if HAVE_PERMLANE
  constexpr unsigned lo = selpart(M, 0);
  constexpr unsigned hi = selpart(M, 8);
  unsigned u = __float_as_uint(x);
  u = (unsigned)__builtin_amdgcn_permlane16((int)u, (int)u, (int)lo, (int)hi, false, false);
  return __uint_as_float(u);
#else
  return __shfl_xor(x, M, 32);
#endif
}

__device__ __forceinline__ void redmax8(float t[8]) {
#pragma unroll
  for (int v = 0; v < 8; ++v) t[v] = fmaxf(t[v], xlane16<1>(t[v]));
#pragma unroll
  for (int v = 0; v < 8; ++v) t[v] = fmaxf(t[v], xlane16<2>(t[v]));
#pragma unroll
  for (int v = 0; v < 8; ++v) t[v] = fmaxf(t[v], xlane16<4>(t[v]));
#pragma unroll
  for (int v = 0; v < 8; ++v) t[v] = fmaxf(t[v], xlane16<8>(t[v]));
}

__device__ __forceinline__ void redsum8(float t[8]) {
#pragma unroll
  for (int v = 0; v < 8; ++v) t[v] += xlane16<1>(t[v]);
#pragma unroll
  for (int v = 0; v < 8; ++v) t[v] += xlane16<2>(t[v]);
#pragma unroll
  for (int v = 0; v < 8; ++v) t[v] += xlane16<4>(t[v]);
#pragma unroll
  for (int v = 0; v < 8; ++v) t[v] += xlane16<8>(t[v]);
}

#if HAVE_TDM
// Build a 2D TDM descriptor (D# groups 0/1 per CDNA5 ISA §8) and issue the load.
// 6-arg builtin form: (g0:v4u, g1:v8i, g2:v4i, g3:v4i, g4:v8i, cpol:i32)
__device__ __forceinline__ void tdm_load_2d(uint32_t lds_byte, uint64_t gaddr,
                                            int ds_code, uint32_t dim0, uint32_t dim1,
                                            uint32_t t0, uint32_t t1, uint64_t stride0) {
  v4u g0;
  g0.x = 1u;                                             // count=1, user mode
  g0.y = lds_byte;                                       // lds_addr
  g0.z = (uint32_t)(gaddr & 0xFFFFFFFFull);              // global_addr lo
  g0.w = (uint32_t)((gaddr >> 32) & 0x01FFFFFFull) | (2u << 30);  // hi + type=2
  v8i g1;
  g1[0] = (int)((uint32_t)ds_code << 16);                // data_size
  g1[1] = (int)((dim0 & 0xFFFFu) << 16);                 // tensor_dim0 lo
  g1[2] = (int)((dim0 >> 16) | ((dim1 & 0xFFFFu) << 16));// dim0 hi | dim1 lo
  g1[3] = (int)((dim1 >> 16) | (t0 << 16));              // dim1 hi | tile_dim0
  g1[4] = (int)(t1 & 0xFFFFu);                           // tile_dim1 (tile_dim2=0)
  g1[5] = (int)(uint32_t)(stride0 & 0xFFFFFFFFull);      // tensor_dim0_stride lo
  g1[6] = (int)(uint32_t)((stride0 >> 32) & 0xFFFFull);  // stride hi (dim1_stride=0)
  g1[7] = 0;
  v4i z4 = {0, 0, 0, 0};
  v8i z8 = {0, 0, 0, 0, 0, 0, 0, 0};
  __builtin_amdgcn_tensor_load_to_lds(g0, g1, z4, z4, z8, 0);
}
#endif

// ---------------- Kernel 0: bf16 copy of x, pack w3 B-fragments, zero stats ----
__global__ void prep_kernel(const float* __restrict__ x, const float* __restrict__ w3,
                            bf16_t* __restrict__ xbf, bf16_t* __restrict__ w3frag,
                            float* __restrict__ gstats) {
  int idx = blockIdx.x * blockDim.x + threadIdx.x;
  if (idx < NCS_) xbf[idx] = (bf16_t)x[idx];
  if (idx < 4096) {
    // frag fi = kc*4+ot, storage [fi][lane][elem]; B layout: lane=col+16*((k/8)&1), e=(k%8)+8*(k/16)
    int fi = idx >> 9, lane = (idx >> 4) & 31, e = idx & 15;
    int kc = fi >> 2, ot = fi & 3;
    int h = lane >> 4, col = lane & 15;
    int c = kc * 32 + 8 * h + (e & 7) + 16 * (e >> 3);
    int o = ot * 16 + col;
    w3frag[idx] = (bf16_t)w3[o * C_ + c];
  }
  if (idx < 128) gstats[idx] = 0.0f;
}

// ---------------- Kernel 1: theta/phi 1x1 convs -> WMMA fragment layouts -------
__global__ void __launch_bounds__(224) qk_kernel(const float* __restrict__ x,
    const float* __restrict__ w1, const float* __restrict__ b1,
    const float* __restrict__ w2, const float* __restrict__ b2,
    bf16_t* __restrict__ qfrag, bf16_t* __restrict__ kfrag) {
  __shared__ float sw[2][CH_][C_];
  __shared__ float sb[2][CH_];
  const int tid = threadIdx.x;
  for (int i = tid; i < CH_ * C_; i += 224) {
    (&sw[0][0][0])[i] = w1[i];
    (&sw[1][0][0])[i] = w2[i];
  }
  if (tid < CH_) { sb[0][tid] = b1[tid]; sb[1][tid] = b2[tid]; }
  __syncthreads();

  const int n = blockIdx.y;
  const int s = blockIdx.x * 224 + tid;
  float xr[C_];
#pragma unroll
  for (int c = 0; c < C_; ++c) xr[c] = x[(size_t)(n * C_ + c) * S_ + s];

  const int tile = s >> 4, m = s & 15;
  const size_t base = (size_t)(n * QT_ + tile) * 512;   // 32 lanes * 16 elems
  for (int i = 0; i < CH_; ++i) {
    float th = sb[0][i], ph = sb[1][i];
#pragma unroll
    for (int c = 0; c < C_; ++c) { th += sw[0][i][c] * xr[c]; ph += sw[1][i][c] * xr[c]; }
    // A-matrix pack (Q): lane = m + 16*((k/8)&1), elem = (k%8) + 8*(k/16)
    int laneA = m + 16 * ((i >> 3) & 1);
    int eA = (i & 7) + 8 * (i >> 4);
    qfrag[base + laneA * 16 + eA] = (bf16_t)(th * SCALE_);
    // B-matrix pack (K): lane = t%16 + 16*((k/8)&1), elem = (k%8) + 8*(k/16)
    int laneB = m + 16 * ((i >> 3) & 1);
    int eB = (i & 7) + 8 * (i >> 4);
    kfrag[base + laneB * 16 + eB] = (bf16_t)ph;
  }
}

// ---------------- Kernel 2: flash attention + w3 epilogue + BN partial stats ---
__global__ void __launch_bounds__(BLK2_) attn_kernel(
    const bf16_t* __restrict__ xbf,    // V: [N][C][S] bf16
    const bf16_t* __restrict__ qfrag,  // [N*QT][32][16]
    const bf16_t* __restrict__ kfrag,  // [N*QT][32][16]
    const bf16_t* __restrict__ w3frag, // [8][32][16]
    const float* __restrict__ b3,
    float* __restrict__ fbuf,          // [N][S][C] f32
    float* __restrict__ gstats) {      // [128]: sum[64], sumsq[64]
  __shared__ __align__(16) unsigned char sK[2][2048];
  __shared__ __align__(16) unsigned char sV[2][4096];
  __shared__ __align__(16) unsigned char sP[WAVES_][2048];
  __shared__ float sStats[128];

  const int tid = threadIdx.x;
  const int wid = tid >> 5;
  const int lane = tid & 31;
  const int col = lane & 15;
  const int half = lane >> 4;
  const int n = blockIdx.y;
  const int qt = blockIdx.x * WAVES_ + wid;

  if (tid < 128) sStats[tid] = 0.0f;

  FragU qa;  // persistent Q A-fragment
  {
    const uint4* qp = (const uint4*)(qfrag + ((size_t)(n * QT_ + qt) * 32 + lane) * 16);
    qa.u[0] = qp[0]; qa.u[1] = qp[1];
  }

  v8f acc[4];
  float mrow[8], lrow[8];
#pragma unroll
  for (int v = 0; v < 8; ++v) { mrow[v] = -3.0e38f; lrow[v] = 0.0f; }
#pragma unroll
  for (int ct = 0; ct < 4; ++ct) acc[ct] = (v8f){};

#if HAVE_TDM
  if (wid == 0) {  // prologue: chunk 0 into buffer 0
    tdm_load_2d((uint32_t)(uintptr_t)&sK[0][0],
                (uint64_t)(uintptr_t)(kfrag + (size_t)(n * QT_) * 512),
                3, 256, 1, 256, 1, 256);                      // 2KB, 8B elements
    tdm_load_2d((uint32_t)(uintptr_t)&sV[0][0],
                (uint64_t)(uintptr_t)(xbf + (size_t)n * C_ * S_),
                1, S_, C_, 32, C_, S_);                       // 64x32 bf16 tile
  }
#endif

  for (int it = 0; it < NKCH_; ++it) {
    const int bi = it & 1;
#if HAVE_TDM
    if (wid == 0) {
      if (it + 1 < NKCH_) {
        const int nb = (it + 1) & 1;
        tdm_load_2d((uint32_t)(uintptr_t)&sK[nb][0],
                    (uint64_t)(uintptr_t)(kfrag + ((size_t)(n * QT_) + (size_t)(it + 1) * 2) * 512),
                    3, 256, 1, 256, 1, 256);
        tdm_load_2d((uint32_t)(uintptr_t)&sV[nb][0],
                    (uint64_t)(uintptr_t)(xbf + (size_t)n * C_ * S_ + (size_t)(it + 1) * 32),
                    1, S_, C_, 32, C_, S_);
        __builtin_amdgcn_s_wait_tensorcnt(2);   // chunk `it` landed
      } else {
        __builtin_amdgcn_s_wait_tensorcnt(0);
      }
    }
#else
    {  // cooperative fallback staging
      uint4* dk = (uint4*)&sK[bi][0];
      const uint4* gk = (const uint4*)(kfrag + ((size_t)(n * QT_) + (size_t)it * 2) * 512);
      for (int i = tid; i < 128; i += BLK2_) dk[i] = gk[i];
      uint4* dv = (uint4*)&sV[bi][0];
      const bf16_t* gv = xbf + (size_t)n * C_ * S_ + (size_t)it * 32;
      for (int i = tid; i < 256; i += BLK2_) {
        int r = i >> 2, q = i & 3;
        dv[i] = ((const uint4*)(gv + (size_t)r * S_))[q];
      }
    }
#endif
    __syncthreads();

    const bf16_t* kls = (const bf16_t*)&sK[bi][0];
    const unsigned char* vls = &sV[bi][0];

    FragU kf0, kf1;
    {
      const uint4* kp0 = (const uint4*)kls + lane * 2;
      kf0.u[0] = kp0[0]; kf0.u[1] = kp0[1];
      const uint4* kp1 = (const uint4*)(kls + 512) + lane * 2;
      kf1.u[0] = kp1[0]; kf1.u[1] = kp1[1];
    }
    v8f zc = {};
    v8f s0 = __builtin_amdgcn_wmma_f32_16x16x32_bf16(false, qa.v, false, kf0.v, (short)0, zc, false, false);
    v8f s1 = __builtin_amdgcn_wmma_f32_16x16x32_bf16(false, qa.v, false, kf1.v, (short)0, zc, false, false);

    // row-max across the 16 lanes of each half (rows live per-VGPR)
    float tm[8];
#pragma unroll
    for (int v = 0; v < 8; ++v) tm[v] = fmaxf(s0[v], s1[v]);
    redmax8(tm);

    float p0[8], p1[8], rs[8];
#pragma unroll
    for (int v = 0; v < 8; ++v) {
      float mn = fmaxf(mrow[v], tm[v]);
      float fac = __expf(mrow[v] - mn);
      mrow[v] = mn;
      lrow[v] *= fac;
      acc[0][v] *= fac; acc[1][v] *= fac; acc[2][v] *= fac; acc[3][v] *= fac;
      p0[v] = __expf(s0[v] - mn);
      p1[v] = __expf(s1[v] - mn);
      rs[v] = p0[v] + p1[v];
    }
    redsum8(rs);
#pragma unroll
    for (int v = 0; v < 8; ++v) lrow[v] += rs[v];

    // P (C-layout) -> LDS (16 rows x 32 cols bf16) -> A-fragment
    bf16_t* pls = (bf16_t*)&sP[wid][0];
#pragma unroll
    for (int v = 0; v < 8; ++v) {
      int row = v + 8 * half;
      pls[row * 32 + col]      = (bf16_t)p0[v];
      pls[row * 32 + 16 + col] = (bf16_t)p1[v];
    }
    FragU pf;
    {
      const unsigned char* pb = (const unsigned char*)pls + col * 64 + half * 16;
      pf.u[0] = *(const uint4*)(pb);
      pf.u[1] = *(const uint4*)(pb + 32);
    }
    // P * V: 4 channel tiles; V B-fragment = keys 8h..8h+7 and 16+8h..16+8h+7 of row c
#pragma unroll
    for (int ct = 0; ct < 4; ++ct) {
      FragU vf;
      const unsigned char* vb = vls + (ct * 16 + col) * 64 + half * 16;
      vf.u[0] = *(const uint4*)(vb);
      vf.u[1] = *(const uint4*)(vb + 32);
      acc[ct] = __builtin_amdgcn_wmma_f32_16x16x32_bf16(false, pf.v, false, vf.v, (short)0, acc[ct], false, false);
    }
    __syncthreads();
  }

  // ------- epilogue: normalize, apply w3 via WMMA, bias, store, stats -------
  float inv[8];
#pragma unroll
  for (int v = 0; v < 8; ++v) inv[v] = 1.0f / lrow[v];
#pragma unroll
  for (int ct = 0; ct < 4; ++ct)
#pragma unroll
    for (int v = 0; v < 8; ++v) acc[ct][v] *= inv[v];

  bf16_t* fls = (bf16_t*)&sP[wid][0];   // reuse staging: 16 rows x 64 cols bf16
#pragma unroll
  for (int ct = 0; ct < 4; ++ct)
#pragma unroll
    for (int v = 0; v < 8; ++v) {
      int row = v + 8 * half;
      fls[row * 64 + ct * 16 + col] = (bf16_t)acc[ct][v];
    }
  FragU fa0, fa1;
  {
    const unsigned char* fb = (const unsigned char*)fls + col * 128 + half * 16;
    fa0.u[0] = *(const uint4*)(fb);
    fa0.u[1] = *(const uint4*)(fb + 32);
    fa1.u[0] = *(const uint4*)(fb + 64);
    fa1.u[1] = *(const uint4*)(fb + 96);
  }
  v8f outv[4];
#pragma unroll
  for (int ct = 0; ct < 4; ++ct) {
    FragU w0, w1;
    const uint4* wp0 = (const uint4*)(w3frag + ((size_t)(0 * 4 + ct) * 32 + lane) * 16);
    w0.u[0] = wp0[0]; w0.u[1] = wp0[1];
    const uint4* wp1 = (const uint4*)(w3frag + ((size_t)(1 * 4 + ct) * 32 + lane) * 16);
    w1.u[0] = wp1[0]; w1.u[1] = wp1[1];
    v8f zc2 = {};
    v8f t = __builtin_amdgcn_wmma_f32_16x16x32_bf16(false, fa0.v, false, w0.v, (short)0, zc2, false, false);
    outv[ct] = __builtin_amdgcn_wmma_f32_16x16x32_bf16(false, fa1.v, false, w1.v, (short)0, t, false, false);
  }
#pragma unroll
  for (int ct = 0; ct < 4; ++ct) {
    float bv = b3[ct * 16 + col];
    float psum = 0.0f, psq = 0.0f;
#pragma unroll
    for (int v = 0; v < 8; ++v) {
      float val = outv[ct][v] + bv;
      int srow = qt * 16 + v + 8 * half;
      fbuf[((size_t)n * S_ + srow) * C_ + ct * 16 + col] = val;
      psum += val; psq += val * val;
    }
    atomicAdd(&sStats[ct * 16 + col], psum);        // ds_add_f32
    atomicAdd(&sStats[64 + ct * 16 + col], psq);
  }
  __syncthreads();
  if (tid < 128) atomicAdd(&gstats[tid], sStats[tid]);
}

// ---------------- Kernel 3: BatchNorm finalize + residual ---------------------
__global__ void bn_kernel(const float* __restrict__ x, const float* __restrict__ fbuf,
                          const float* __restrict__ gstats,
                          const float* __restrict__ gamma, const float* __restrict__ beta,
                          float* __restrict__ out) {
  int idx = blockIdx.x * blockDim.x + threadIdx.x;
  if (idx >= NCS_) return;
  int c = (idx / S_) & (C_ - 1);
  int n = idx / (C_ * S_);
  int s = idx % S_;
  const float invM = 1.0f / (float)(N_ * S_);
  float mean = gstats[c] * invM;
  float var = gstats[64 + c] * invM - mean * mean;
  float f = fbuf[((size_t)n * S_ + s) * C_ + c];
  float y = (f - mean) * rsqrtf(var + BN_EPS_) * gamma[c] + beta[c];
  out[idx] = x[idx] + y;
}

extern "C" void kernel_launch(void* const* d_in, const int* in_sizes, int n_in,
                              void* d_out, int out_size, void* d_ws, size_t ws_size,
                              hipStream_t stream) {
  (void)in_sizes; (void)n_in; (void)out_size; (void)ws_size;
  const float* x     = (const float*)d_in[0];
  const float* w1    = (const float*)d_in[1];
  const float* b1    = (const float*)d_in[2];
  const float* w2    = (const float*)d_in[3];
  const float* b2    = (const float*)d_in[4];
  const float* w3    = (const float*)d_in[5];
  const float* b3    = (const float*)d_in[6];
  const float* gamma = (const float*)d_in[7];
  const float* beta  = (const float*)d_in[8];

  char* ws = (char*)d_ws;
  size_t off = 0;
  auto alloc = [&](size_t bytes) {
    char* p = ws + off;
    off += (bytes + 1023) & ~(size_t)1023;
    return p;
  };
  bf16_t* xbf    = (bf16_t*)alloc((size_t)NCS_ * 2);
  bf16_t* qfrag  = (bf16_t*)alloc((size_t)NT_ * 512 * 2);
  bf16_t* kfrag  = (bf16_t*)alloc((size_t)NT_ * 512 * 2);
  bf16_t* w3frag = (bf16_t*)alloc(4096 * 2);
  float*  fbuf   = (float*)alloc((size_t)NCS_ * 4);
  float*  gstats = (float*)alloc(128 * 4);

  prep_kernel<<<(NCS_ + 255) / 256, 256, 0, stream>>>(x, w3, xbf, w3frag, gstats);
  dim3 g1(S_ / 224, N_);
  qk_kernel<<<g1, 224, 0, stream>>>(x, w1, b1, w2, b2, qfrag, kfrag);
  dim3 g2(QT_ / WAVES_, N_);
  attn_kernel<<<g2, BLK2_, 0, stream>>>(xbf, qfrag, kfrag, w3frag, b3, fbuf, gstats);
  bn_kernel<<<(NCS_ + 255) / 256, 256, 0, stream>>>(x, fbuf, gstats, gamma, beta, (float*)d_out);
}
